// FPN2MLPFeatureExtractorContext_44126493999180
// MI455X (gfx1250) — compile-verified, matrix operand order checked by reading, not run on hardware
//
#include <hip/hip_runtime.h>

// ---------------------------------------------------------------------------
// Types for CDNA5 WMMA (gfx1250, wave32):  D(16x16 f32) = A(16x32 f16) x B(32x16 f16) + C
// ---------------------------------------------------------------------------
typedef __attribute__((ext_vector_type(16))) _Float16 v16h;
typedef __attribute__((ext_vector_type(4)))  _Float16 v4h;
typedef __attribute__((ext_vector_type(4)))  float    v4f;
typedef __attribute__((ext_vector_type(8)))  float    v8f;

#define WMMA_F16(a, b, c) \
    __builtin_amdgcn_wmma_f32_16x16x32_f16(false, (a), false, (b), (short)0, (c), false, false)

// ---------------------------------------------------------------------------
// Workspace layout (in _Float16 elements). Total = 88,080,384 halves = 168 MiB.
// ---------------------------------------------------------------------------
static constexpr size_t WBRH   = 0;                    // w1,w2,w3,w6 contiguous: 4 * 256*256
static constexpr size_t WCATH  = WBRH   + 4 * 65536;   // 256*1024
static constexpr size_t FC6CH  = WCATH  + 262144;      // 1024*12544
static constexpr size_t FC7CH  = FC6CH  + 12845056;    // 1024*1024
static constexpr size_t FC6RH  = FC7CH  + 1048576;     // 1024*12544
static constexpr size_t FC7RH  = FC6RH  + 12845056;    // 1024*1024
static constexpr size_t FH     = FC7RH  + 1048576;     // 4096*12544  (fused feature, fp16)
static constexpr size_t HCH    = FH     + 51380224;    // 4096*1024   (fc6c hidden, fp16)
static constexpr size_t HRH    = HCH    + 4194304;     // 4096*1024   (fc6r hidden, fp16)

// ---------------------------------------------------------------------------
// fp32 -> fp16 conversion, 4-wide (all tensor sizes are multiples of 4)
// ---------------------------------------------------------------------------
__global__ __launch_bounds__(256) void convert_f32_f16(const float* __restrict__ s,
                                                       _Float16* __restrict__ d, int n4) {
    const v4f* __restrict__ s4 = (const v4f*)s;
    v4h* __restrict__ d4 = (v4h*)d;
    for (int i = blockIdx.x * blockDim.x + threadIdx.x; i < n4; i += gridDim.x * blockDim.x) {
        v4f v = s4[i];
        v4h h;
        h.x = (_Float16)v.x; h.y = (_Float16)v.y;
        h.z = (_Float16)v.z; h.w = (_Float16)v.w;
        d4[i] = h;
    }
}

// ---------------------------------------------------------------------------
// Branch kernel: one block (256 threads = 8 waves) per image.
//   - pools / attention / softmax in VALU (tiny work)
//   - stage A: relu(w_i @ o_i)      via WMMA  -> LDS cat segment (fp16)
//   - stage B: w_cat_seg @ cat_seg  via WMMA, accumulated over 4 branches
//   - epilogue: f = x + sum  -> LDS staging -> coalesced global fp16
// ---------------------------------------------------------------------------
__global__ __launch_bounds__(256) void branch_fuse(const float* __restrict__ x,
                                                   const _Float16* __restrict__ wbase,  // 4 x [256][256]
                                                   const _Float16* __restrict__ wcath,  // [256][1024]
                                                   _Float16* __restrict__ f_h) {
    __shared__ float    xs[256 * 49];       // x[b] as [C][49]           (50176 B)
    __shared__ float    pooled[256 * 36];   // [C][P<=36]                (36864 B)
    __shared__ float    attn[49 * 36];      // logits / softmax          ( 7056 B)
    __shared__ _Float16 ot[64 * 256];       // o^T [n_pad][C]; reused as f staging (32768 B)
    __shared__ _Float16 catseg[64 * 256];   // relu(conv)^T [n_pad][C]   (32768 B)

    const int b    = blockIdx.x;
    const int tid  = threadIdx.x;
    const int lane = tid & 31;
    const int wave = tid >> 5;

    // load x[b] -> LDS
    for (int i = tid; i < 256 * 49; i += 256) xs[i] = x[(size_t)b * 12544 + i];
    // zero padding rows of o^T (n = 49..63) once
    for (int i = tid; i < 15 * 256; i += 256) ot[49 * 256 + i] = (_Float16)0.f;
    __syncthreads();

    const int osz[4] = {1, 2, 3, 6};
    const int koff = (lane >> 4) << 4;   // 0 or 16: K-half selector for fragments

    v8f accB[8];
    #pragma unroll
    for (int t = 0; t < 8; ++t) accB[t] = {};

    for (int bi = 0; bi < 4; ++bi) {
        const int o = osz[bi];
        const int P = o * o;

        // ---- adaptive avg pool: pooled[c][p] ----
        for (int idx = tid; idx < 256 * P; idx += 256) {
            int c = idx / P, p = idx - c * P;
            int py = p / o, px = p - py * o;
            int sy = (py * 7) / o, ey = ((py + 1) * 7 + o - 1) / o;
            int sx = (px * 7) / o, ex = ((px + 1) * 7 + o - 1) / o;
            float s = 0.f;
            for (int yy = sy; yy < ey; ++yy)
                for (int xx = sx; xx < ex; ++xx) s += xs[c * 49 + yy * 7 + xx];
            pooled[c * 36 + p] = s / (float)((ey - sy) * (ex - sx));
        }
        __syncthreads();

        // ---- attention logits: attn[n][p] = sum_c x[c][n] * pooled[c][p] ----
        for (int idx = tid; idx < 49 * P; idx += 256) {
            int n = idx / P, p = idx - n * P;
            float s = 0.f;
            for (int c = 0; c < 256; ++c) s += xs[c * 49 + n] * pooled[c * 36 + p];
            attn[n * 36 + p] = s;
        }
        __syncthreads();

        // ---- softmax over p (per n) ----
        if (tid < 49) {
            float m = attn[tid * 36];
            for (int p = 1; p < P; ++p) m = fmaxf(m, attn[tid * 36 + p]);
            float sum = 0.f;
            for (int p = 0; p < P; ++p) {
                float e = expf(attn[tid * 36 + p] - m);
                attn[tid * 36 + p] = e;
                sum += e;
            }
            float inv = 1.f / sum;
            for (int p = 0; p < P; ++p) attn[tid * 36 + p] *= inv;
        }
        __syncthreads();

        // ---- o^T[n][c] = sum_p pooled[c][p] * attn[n][p]  (fp16) ----
        for (int idx = tid; idx < 49 * 256; idx += 256) {
            int n = idx >> 8, c = idx & 255;
            float s = 0.f;
            for (int p = 0; p < P; ++p) s += pooled[c * 36 + p] * attn[n * 36 + p];
            ot[n * 256 + c] = (_Float16)s;
        }
        __syncthreads();

        // ---- stage A: catseg = relu(w_i @ o)  via WMMA ----
        // 64 tiles (16 M-tiles x 4 N-tiles) over 8 waves; wave w: mt in {2w,2w+1}, nt 0..3
        const _Float16* __restrict__ wh = wbase + (size_t)bi * 65536;  // direct global provenance
        #pragma unroll
        for (int t = 0; t < 8; ++t) {
            const int mt = wave * 2 + (t >> 2);
            const int nt = t & 3;
            const int row  = mt * 16 + (lane & 15);   // output channel row of w_i
            const int ncol = nt * 16 + (lane & 15);   // spatial position n
            v8f acc = {};
            for (int k0 = 0; k0 < 256; k0 += 32) {
                v16h a = *(const v16h*)(wh + (size_t)row * 256 + k0 + koff);
                v16h bf = *(const v16h*)(ot + ncol * 256 + k0 + koff);
                acc = WMMA_F16(a, bf, acc);
            }
            const int mbase = mt * 16 + ((lane >> 4) << 3);
            #pragma unroll
            for (int i = 0; i < 8; ++i) {
                float v = acc[i];
                catseg[ncol * 256 + mbase + i] = (_Float16)(v > 0.f ? v : 0.f);
            }
        }
        __syncthreads();

        // ---- stage B: accB += w_cat[:, seg] @ catseg  via WMMA ----
        #pragma unroll
        for (int t = 0; t < 8; ++t) {
            const int mt = wave * 2 + (t >> 2);
            const int nt = t & 3;
            const int row  = mt * 16 + (lane & 15);
            const int ncol = nt * 16 + (lane & 15);
            for (int k0 = 0; k0 < 256; k0 += 32) {
                v16h a = *(const v16h*)(wcath + (size_t)row * 1024 + bi * 256 + k0 + koff);
                v16h bf = *(const v16h*)(catseg + ncol * 256 + k0 + koff);
                accB[t] = WMMA_F16(a, bf, accB[t]);
            }
        }
        __syncthreads();
    }

    // ---- epilogue: f = x + conv_cat(cat) -> LDS staging (reuse ot) ----
    _Float16* __restrict__ fbuf = ot;  // 16384 halves >= 12544
    #pragma unroll
    for (int t = 0; t < 8; ++t) {
        const int mt = wave * 2 + (t >> 2);
        const int nt = t & 3;
        const int n = nt * 16 + (lane & 15);
        const int mbase = mt * 16 + ((lane >> 4) << 3);
        if (n < 49) {
            #pragma unroll
            for (int i = 0; i < 8; ++i) {
                const int c = mbase + i;
                fbuf[c * 49 + n] = (_Float16)(xs[c * 49 + n] + accB[t][i]);
            }
        }
    }
    __syncthreads();

    // coalesced stream-out: 12544 halves = 6272 dwords
    const unsigned int* __restrict__ src = (const unsigned int*)fbuf;
    unsigned int* __restrict__ dst = (unsigned int*)(f_h + (size_t)b * 12544);
    for (int i = tid; i < 6272; i += 256) dst[i] = src[i];
}

// ---------------------------------------------------------------------------
// WMMA GEMM:  out[M,N] = act( A[M,K](f16) @ Bw[N,K]^T(f16) + bias[N] )
// 128x128 block tile, 8 waves, each wave 64x32 (4x2 wmma tiles), K-step 32.
// Double-buffered LDS: prefetch tile k+1 while WMMAs consume tile k.
// ---------------------------------------------------------------------------
__global__ __launch_bounds__(256) void gemm_wmma(const _Float16* __restrict__ A,
                                                 const _Float16* __restrict__ Bw,
                                                 const float* __restrict__ bias,
                                                 float* __restrict__ out32,
                                                 _Float16* __restrict__ out16,
                                                 int M, int N, int K, int relu) {
    __shared__ _Float16 As[2][128 * 48];  // stride 48 halves (96 B): 32B-aligned, bank-shifted
    __shared__ _Float16 Bs[2][128 * 48];

    const int tid  = threadIdx.x;
    const int lane = tid & 31;
    const int wave = tid >> 5;
    const int wm = wave & 1;              // 2 wave-rows x 4 wave-cols
    const int wn = wave >> 1;

    const int m0 = blockIdx.y * 128;
    const int n0 = blockIdx.x * 128;

    const int lrow = tid >> 1;            // 0..127
    const int lseg = (tid & 1) << 4;      // 0 or 16 halves
    const int koff = (lane >> 4) << 4;
    const int loff = lrow * 48 + lseg;

    v8f acc[8];
    #pragma unroll
    for (int t = 0; t < 8; ++t) acc[t] = {};

    const _Float16* __restrict__ ga = A  + (size_t)(m0 + lrow) * K + lseg;
    const _Float16* __restrict__ gb = Bw + (size_t)(n0 + lrow) * K + lseg;

    // preload K-tile 0
    *(v16h*)(As[0] + loff) = *(const v16h*)(ga);
    *(v16h*)(Bs[0] + loff) = *(const v16h*)(gb);
    __syncthreads();

    int cur = 0;
    for (int k0 = 0; k0 < K; k0 += 32, cur ^= 1) {
        if (k0 + 32 < K) {
            // fill the other buffer while this one is consumed
            *(v16h*)(As[cur ^ 1] + loff) = *(const v16h*)(ga + k0 + 32);
            *(v16h*)(Bs[cur ^ 1] + loff) = *(const v16h*)(gb + k0 + 32);
            if (k0 + 64 < K) {
                __builtin_prefetch(ga + k0 + 64, 0, 0);
                __builtin_prefetch(gb + k0 + 64, 0, 0);
            }
        }
        #pragma unroll
        for (int ms = 0; ms < 4; ++ms) {
            const int row = wm * 64 + ms * 16 + (lane & 15);
            v16h a = *(const v16h*)(As[cur] + row * 48 + koff);
            #pragma unroll
            for (int ns = 0; ns < 2; ++ns) {
                const int col = wn * 32 + ns * 16 + (lane & 15);
                v16h bf = *(const v16h*)(Bs[cur] + col * 48 + koff);
                acc[ms * 2 + ns] = WMMA_F16(a, bf, acc[ms * 2 + ns]);
            }
        }
        __syncthreads();  // next buffer fully written AND current fully consumed
    }

    #pragma unroll
    for (int ms = 0; ms < 4; ++ms) {
        #pragma unroll
        for (int ns = 0; ns < 2; ++ns) {
            const int gm_base = m0 + wm * 64 + ms * 16 + ((lane >> 4) << 3);
            const int gn      = n0 + wn * 32 + ns * 16 + (lane & 15);
            const float bv = bias[gn];
            #pragma unroll
            for (int i = 0; i < 8; ++i) {
                float v = acc[ms * 2 + ns][i] + bv;
                if (relu) v = v > 0.f ? v : 0.f;
                const size_t oi = (size_t)(gm_base + i) * N + gn;
                if (out32) out32[oi] = v;
                if (out16) out16[oi] = (_Float16)v;
            }
        }
    }
}

// ---------------------------------------------------------------------------
extern "C" void kernel_launch(void* const* d_in, const int* in_sizes, int n_in,
                              void* d_out, int out_size, void* d_ws, size_t ws_size,
                              hipStream_t stream) {
    const float* x       = (const float*)d_in[0];
    const float* w1      = (const float*)d_in[1];
    const float* w2      = (const float*)d_in[2];
    const float* w3      = (const float*)d_in[3];
    const float* w6      = (const float*)d_in[4];
    const float* w_cat   = (const float*)d_in[5];
    const float* fc6c_w  = (const float*)d_in[6];
    const float* fc6c_b  = (const float*)d_in[7];
    const float* fc7c_w  = (const float*)d_in[8];
    const float* fc7c_b  = (const float*)d_in[9];
    const float* fc6r_w  = (const float*)d_in[10];
    const float* fc6r_b  = (const float*)d_in[11];
    const float* fc7r_w  = (const float*)d_in[12];
    const float* fc7r_b  = (const float*)d_in[13];

    _Float16* ws = (_Float16*)d_ws;
    float* out = (float*)d_out;

    auto cvt = [&](const float* s, size_t off, int n) {
        int n4 = n >> 2;
        int blocks = (n4 + 255) / 256;
        if (blocks > 4096) blocks = 4096;
        convert_f32_f16<<<blocks, 256, 0, stream>>>(s, ws + off, n4);
    };
    cvt(w1,     WBRH + 0 * 65536, 65536);
    cvt(w2,     WBRH + 1 * 65536, 65536);
    cvt(w3,     WBRH + 2 * 65536, 65536);
    cvt(w6,     WBRH + 3 * 65536, 65536);
    cvt(w_cat,  WCATH, 262144);
    cvt(fc6c_w, FC6CH, 12845056);
    cvt(fc7c_w, FC7CH, 1048576);
    cvt(fc6r_w, FC6RH, 12845056);
    cvt(fc7r_w, FC7RH, 1048576);

    // Fused attention branches + concat conv + residual -> f (fp16)
    branch_fuse<<<4096, 256, 0, stream>>>(x, ws + WBRH, ws + WCATH, ws + FH);

    // MLP heads (all WMMA): grid = (N/128, M/128)
    dim3 g6(1024 / 128, 4096 / 128);
    gemm_wmma<<<g6, 256, 0, stream>>>(ws + FH,  ws + FC6CH, fc6c_b, nullptr, ws + HCH,
                                      4096, 1024, 12544, 1);
    gemm_wmma<<<g6, 256, 0, stream>>>(ws + HCH, ws + FC7CH, fc7c_b, out, nullptr,
                                      4096, 1024, 1024, 0);
    gemm_wmma<<<g6, 256, 0, stream>>>(ws + FH,  ws + FC6RH, fc6r_b, nullptr, ws + HRH,
                                      4096, 1024, 12544, 1);
    gemm_wmma<<<g6, 256, 0, stream>>>(ws + HRH, ws + FC7RH, fc7r_b, out + 4096 * 1024, nullptr,
                                      4096, 1024, 1024, 1);
}
